// SpatialChannelAttention_87600152969567
// MI455X (gfx1250) — matrix operand order repaired and checked by reading, not compiled, not run on hardware
//
#include <hip/hip_runtime.h>

typedef __attribute__((ext_vector_type(16))) _Float16 v16h;
typedef __attribute__((ext_vector_type(8)))  float    v8f;

#define N_B      8
#define L_SEQ    4096
#define H_HEADS  8
#define D_SM     32
#define NH       (N_B * H_HEADS)
#define SEG      8
#define SEG_LEN  (L_SEQ / SEG)   // 512

__device__ __forceinline__ float elu1(float x) {
  // elu(x) + 1
  return x > 0.0f ? x + 1.0f : __expf(x);
}

// ---------------------------------------------------------------------------
// Kernel 1: linear attention statistics.
// Per (n, h, seg): partial KV[32][32] = sum_s elu1(K[s,d]) * V[s,v]
//                  partial Ksum[32]   = sum_s elu1(K[s,d])
// (The /v_len and *v_len in the reference cancel exactly, so V is used raw.)
// ---------------------------------------------------------------------------
__global__ __launch_bounds__(256)
void lin_stats_kernel(const float* __restrict__ k,
                      const float* __restrict__ v,
                      float* __restrict__ kvp,   // [NH*SEG][1024]
                      float* __restrict__ ksp) { // [NH*SEG][32]
  __shared__ float Kc[64][32];
  __shared__ float Vc[64][32];
  const int seg = blockIdx.x, h = blockIdx.y, n = blockIdx.z;
  const int t   = threadIdx.x;
  const int rr  = t >> 5;     // 0..7  (uniform within a wave32)
  const int col = t & 31;     // lane
  float kv[4] = {0.f, 0.f, 0.f, 0.f};
  float ks[4] = {0.f, 0.f, 0.f, 0.f};
  const size_t base = ((size_t)n * L_SEQ) * (H_HEADS * D_SM) + (size_t)h * D_SM;

  for (int chunk = 0; chunk < SEG_LEN / 64; ++chunk) {
    const int s0 = seg * SEG_LEN + chunk * 64;
    __syncthreads();
    for (int e = t; e < 64 * 32; e += 256) {
      const int row = e >> 5, c = e & 31;
      const size_t g = base + (size_t)(s0 + row) * (H_HEADS * D_SM) + c;
      Kc[row][c] = elu1(k[g]);
      Vc[row][c] = v[g];
    }
    __syncthreads();
    for (int s = 0; s < 64; ++s) {
      const float vv = Vc[s][col];
#pragma unroll
      for (int a = 0; a < 4; ++a) {
        const float kk = Kc[s][rr + 8 * a];   // uniform -> LDS broadcast
        kv[a] = fmaf(kk, vv, kv[a]);
        ks[a] += kk;
      }
    }
  }
  const int slot = (n * H_HEADS + h) * SEG + seg;
#pragma unroll
  for (int a = 0; a < 4; ++a)
    kvp[(size_t)slot * 1024 + (rr + 8 * a) * 32 + col] = kv[a];
  if (col == 0) {
#pragma unroll
    for (int a = 0; a < 4; ++a) ksp[slot * 32 + rr + 8 * a] = ks[a];
  }
}

// ---------------------------------------------------------------------------
// Kernel 2: channel attention via WMMA f16.  One block per (n, h1) slice.
// The channel view of a slice is 32768 contiguous floats: elem(r, c) at
// base + c*256 + r  (r in [0,256) rows, c in [0,128) head dim).
// Writes out with "=" (covers every element of d_out exactly once).
// ---------------------------------------------------------------------------
__global__ __launch_bounds__(512)
void channel_attn_kernel(const float* __restrict__ q,
                         const float* __restrict__ k,
                         const float* __restrict__ v,
                         const float* __restrict__ alpha_p,
                         float* __restrict__ out) {
  extern __shared__ _Float16 smem[];
  _Float16* kfr  = smem;            // 64 B-frags * 512 halfs = 64 KB
  _Float16* vfr  = smem + 32768;    // 64 B-frags * 512 halfs = 64 KB
  _Float16* pbuf = smem + 65536;    // 16 waves * 8 A-frags * 512 = 128 KB

  const int h1 = blockIdx.x, n = blockIdx.y;
  const int tid  = threadIdx.x;
  const int lane = tid & 31;
  const int wave = tid >> 5;
  const bool hiHalf = lane >= 16;
  const size_t cbase = ((size_t)(n * L_SEQ + h1 * 128)) * 256;
  const float beta = 1.0f - alpha_p[0];
  const float scale = 0.0883883476483184405f;  // 1/sqrt(128)

  // --- stage K into B-fragment layout: frag = sTile*4 + cChunk
  //     B 32x16: lanes 0-15 hold K {0..7,16..23}, lanes 16-31 hold K {8..15,24..31}.
  const float* kg = k + cbase;
  for (int e = tid; e < 32768; e += 512) {
    const int c = e >> 8, s = e & 255;          // contraction c, column s
    const int frag  = (s >> 4) * 4 + (c >> 5);
    const int lane2 = (s & 15) + (((c & 15) >= 8) ? 16 : 0);
    const int p     = (c & 7) + (((c & 31) >= 16) ? 8 : 0);
    kfr[frag * 512 + lane2 * 16 + p] = (_Float16)kg[e];
  }
  // --- stage V into B-fragment layout: frag = dTile*8 + sChunk (contraction s)
  const float* vg = v + cbase;
  for (int e = tid; e < 32768; e += 512) {
    const int d = e >> 8, s = e & 255;
    const int frag  = (d >> 4) * 8 + (s >> 5);
    const int lane2 = (d & 15) + (((s & 15) >= 8) ? 16 : 0);
    const int p     = (s & 7) + (((s & 31) >= 16) ? 8 : 0);
    vfr[frag * 512 + lane2 * 16 + p] = (_Float16)vg[e];
  }
  __syncthreads();

  // --- A fragments of this wave's 16 Q rows (ISA 16-bit A 16x32 layout:
  //     lane<16 holds K {0..7,16..23}; lane>=16 holds K {8..15,24..31})
  const int m0   = wave * 16;
  const int mrow = m0 + (lane & 15);
  const float* qg = q + cbase;
  v16h Aq[4];
#pragma unroll
  for (int cc = 0; cc < 4; ++cc) {
#pragma unroll
    for (int p = 0; p < 16; ++p) {
      int cl = (p < 8) ? p : p + 8;
      if (hiHalf) cl += 8;
      Aq[cc][p] = (_Float16)qg[(size_t)(cc * 32 + cl) * 256 + mrow];
    }
  }

  // --- scores: S(16x256) = Q(16x128) @ K^T, 16 tiles x 4 K-chunks of WMMA
  v8f S[16];
#pragma unroll
  for (int t = 0; t < 16; ++t) {
    v8f acc = {};
#pragma unroll
    for (int cc = 0; cc < 4; ++cc) {
      v16h B = *((const v16h*)(kfr + (t * 4 + cc) * 512 + lane * 16));
      acc = __builtin_amdgcn_wmma_f32_16x16x32_f16(false, Aq[cc], false, B,
                                                   (short)0, acc, false, false);
    }
    S[t] = acc;
  }

  // --- row softmax in C layout (row m = j + 8*hiHalf; N = lane%16 per tile)
  float rowmax[8], rinv[8];
#pragma unroll
  for (int j = 0; j < 8; ++j) {
    float m = -3.0e38f;
#pragma unroll
    for (int t = 0; t < 16; ++t) m = fmaxf(m, S[t][j]);
    m = fmaxf(m, __shfl_xor(m, 1, 32));
    m = fmaxf(m, __shfl_xor(m, 2, 32));
    m = fmaxf(m, __shfl_xor(m, 4, 32));
    m = fmaxf(m, __shfl_xor(m, 8, 32));   // xor<=8 stays inside 16-lane half
    rowmax[j] = m;
  }
#pragma unroll
  for (int t = 0; t < 16; ++t) {
#pragma unroll
    for (int j = 0; j < 8; ++j)
      S[t][j] = __expf((S[t][j] - rowmax[j]) * scale);
  }
#pragma unroll
  for (int j = 0; j < 8; ++j) {
    float s = 0.f;
#pragma unroll
    for (int t = 0; t < 16; ++t) s += S[t][j];
    s += __shfl_xor(s, 1, 32);
    s += __shfl_xor(s, 2, 32);
    s += __shfl_xor(s, 4, 32);
    s += __shfl_xor(s, 8, 32);
    rinv[j] = 1.0f / s;
  }

  // --- store P into wave-private LDS in A-fragment layout (C->A transpose)
  _Float16* pw = pbuf + wave * 4096;
#pragma unroll
  for (int t = 0; t < 16; ++t) {
    const int s     = t * 16 + (lane & 15);
    const int chunk = s >> 5;
    const int sl    = s & 31;
    const int hi2   = ((sl & 15) >= 8) ? 16 : 0;
    const int p     = (sl & 7) + ((sl >= 16) ? 8 : 0);
#pragma unroll
    for (int j = 0; j < 8; ++j) {
      const int m_local = j + (hiHalf ? 8 : 0);
      pw[chunk * 512 + (m_local + hi2) * 16 + p] = (_Float16)S[t][j];
    }
  }
  __asm volatile("s_wait_dscnt 0" ::: "memory");  // wave-private round-trip

  // --- out(16x128) = P(16x256) @ V(256x128): 8 s-chunks x 8 d-tiles of WMMA
  v8f acc[8];
#pragma unroll
  for (int dt = 0; dt < 8; ++dt) acc[dt] = (v8f){};
#pragma unroll
  for (int sc = 0; sc < 8; ++sc) {
    v16h Ap = *((const v16h*)(pw + sc * 512 + lane * 16));
#pragma unroll
    for (int dt = 0; dt < 8; ++dt) {
      v16h Bv = *((const v16h*)(vfr + (dt * 8 + sc) * 512 + lane * 16));
      acc[dt] = __builtin_amdgcn_wmma_f32_16x16x32_f16(false, Ap, false, Bv,
                                                       (short)0, acc[dt],
                                                       false, false);
    }
  }
  // --- normalize + write back through the channel view (out = beta * m2)
  float* og = out + cbase;
#pragma unroll
  for (int dt = 0; dt < 8; ++dt) {
    const int d = dt * 16 + (lane & 15);
#pragma unroll
    for (int j = 0; j < 8; ++j) {
      const int m = m0 + j + (hiHalf ? 8 : 0);
      og[(size_t)d * 256 + m] = beta * acc[dt][j] * rinv[j];
    }
  }
}

// ---------------------------------------------------------------------------
// Kernel 3: linear attention apply.  out += alpha * (Q@KV) / (Q.Ksum + eps).
// Reduces the SEG partials, then each thread handles one sequence position.
// ---------------------------------------------------------------------------
__global__ __launch_bounds__(256)
void lin_apply_kernel(const float* __restrict__ q,
                      const float* __restrict__ kvp,
                      const float* __restrict__ ksp,
                      const float* __restrict__ alpha_p,
                      float* __restrict__ out) {
  __shared__ float KVs[1024];
  __shared__ float KSs[32];
  const int lc = blockIdx.x, h = blockIdx.y, n = blockIdx.z;
  const int t = threadIdx.x;
  const int slot0 = (n * H_HEADS + h) * SEG;
  for (int e = t; e < 1024; e += 256) {
    float s = 0.f;
#pragma unroll
    for (int g = 0; g < SEG; ++g) s += kvp[(size_t)(slot0 + g) * 1024 + e];
    KVs[e] = s;
  }
  if (t < 32) {
    float s = 0.f;
#pragma unroll
    for (int g = 0; g < SEG; ++g) s += ksp[(slot0 + g) * 32 + t];
    KSs[t] = s;
  }
  __syncthreads();

  const float alpha = alpha_p[0];
  const int l = lc * 256 + t;
  const size_t qb = (((size_t)n * L_SEQ + l) * H_HEADS + h) * D_SM;
  float Qf[32];
#pragma unroll
  for (int d4 = 0; d4 < 8; ++d4) {
    const float4 qv = *((const float4*)(q + qb + d4 * 4));
    Qf[d4 * 4 + 0] = elu1(qv.x);
    Qf[d4 * 4 + 1] = elu1(qv.y);
    Qf[d4 * 4 + 2] = elu1(qv.z);
    Qf[d4 * 4 + 3] = elu1(qv.w);
  }
  float z = 0.f;
#pragma unroll
  for (int d = 0; d < 32; ++d) z = fmaf(Qf[d], KSs[d], z);
  z = alpha / (z + 1e-6f);

  float* ob = out + qb;
#pragma unroll
  for (int v0 = 0; v0 < 32; v0 += 4) {
    float ax = 0.f, ay = 0.f, az = 0.f, aw = 0.f;
#pragma unroll
    for (int d = 0; d < 32; ++d) {
      const float4 kv4 = *((const float4*)(KVs + d * 32 + v0));  // broadcast
      ax = fmaf(Qf[d], kv4.x, ax);
      ay = fmaf(Qf[d], kv4.y, ay);
      az = fmaf(Qf[d], kv4.z, az);
      aw = fmaf(Qf[d], kv4.w, aw);
    }
    float4 prev = *((float4*)(ob + v0));
    prev.x = fmaf(z, ax, prev.x);
    prev.y = fmaf(z, ay, prev.y);
    prev.z = fmaf(z, az, prev.z);
    prev.w = fmaf(z, aw, prev.w);
    *((float4*)(ob + v0)) = prev;
  }
}

extern "C" void kernel_launch(void* const* d_in, const int* in_sizes, int n_in,
                              void* d_out, int out_size, void* d_ws, size_t ws_size,
                              hipStream_t stream) {
  const float* q     = (const float*)d_in[0];
  const float* k     = (const float*)d_in[1];
  const float* v_sp  = (const float*)d_in[2];
  const float* v_ch  = (const float*)d_in[3];
  const float* alpha = (const float*)d_in[6];
  float* out = (float*)d_out;

  float* kvp = (float*)d_ws;                       // NH*SEG*1024 floats (2 MB)
  float* ksp = kvp + (size_t)NH * SEG * 1024;      // NH*SEG*32 floats

  // 1) linear-attention partial KV / Ksum (independent of d_out)
  lin_stats_kernel<<<dim3(SEG, H_HEADS, N_B), 256, 0, stream>>>(k, v_sp, kvp, ksp);
  // 2) channel attention: writes out = (1-alpha)*m2  (WMMA path, 256 KB LDS)
  channel_attn_kernel<<<dim3(L_SEQ / 128, N_B), 512, 256 * 1024, stream>>>(
      q, k, v_ch, alpha, out);
  // 3) linear attention: out += alpha*m1 (coalesced float4 RMW)
  lin_apply_kernel<<<dim3(L_SEQ / 256, H_HEADS, N_B), 256, 0, stream>>>(
      q, kvp, ksp, alpha, out);

  (void)in_sizes; (void)n_in; (void)out_size; (void)ws_size;
}
